// Attention_88218628260019
// MI455X (gfx1250) — matrix operand order repaired and checked by reading, not compiled, not run on hardware
//
#include <hip/hip_runtime.h>
#include <math.h>

typedef __attribute__((ext_vector_type(16))) _Float16 v16h;
typedef __attribute__((ext_vector_type(8)))  _Float16 v8h;
typedef __attribute__((ext_vector_type(8)))  float    v8f;

#define NH   32
#define NKV  8
#define DH   128
#define GQ   4      // NH / NKV
#define QT   16     // query rows per workgroup

constexpr int KP = DH + 8;   // LDS pitch (halfs) for K rows (pad kills bank conflicts)
constexpr int VP = 32 + 8;   // LDS pitch (halfs) for Vt/P rows

union AV { v16h v; v8h h[2]; };

// ---- DPP16 16-lane reductions (VALU only; keeps the DS pipe free for B-tiles) ----
// xor1 = quad_perm(1,0,3,2)=0xB1, xor2 = quad_perm(2,3,0,1)=0x4E,
// then row_half_mirror(0x141) and row_mirror(0x140) finish the 16-lane tree.
#define DPPMOV(x, ctrl)                                                         \
  __builtin_bit_cast(float, __builtin_amdgcn_update_dpp(                        \
      __builtin_bit_cast(int, (x)), __builtin_bit_cast(int, (x)), (ctrl), 0xf, \
      0xf, true))

__device__ inline float redmax16(float x) {
  x = fmaxf(x, DPPMOV(x, 0xB1));
  x = fmaxf(x, DPPMOV(x, 0x4E));
  x = fmaxf(x, DPPMOV(x, 0x141));
  x = fmaxf(x, DPPMOV(x, 0x140));
  return x;
}
__device__ inline float redsum16(float x) {
  x += DPPMOV(x, 0xB1);
  x += DPPMOV(x, 0x4E);
  x += DPPMOV(x, 0x141);
  x += DPPMOV(x, 0x140);
  return x;
}

// ---- async global->LDS copy: 4 x b128 per call (ASYNCcnt tracked) ----
// INST_OFFSET is added to BOTH the LDS address and the global address.
__device__ inline void async_cp4(unsigned lds, const _Float16* base, int byteoff) {
  asm volatile(
      "global_load_async_to_lds_b128 %0, %1, %2 offset:0\n\t"
      "global_load_async_to_lds_b128 %0, %1, %2 offset:16\n\t"
      "global_load_async_to_lds_b128 %0, %1, %2 offset:32\n\t"
      "global_load_async_to_lds_b128 %0, %1, %2 offset:48"
      :: "v"(lds), "v"(byteoff), "s"(base) : "memory");
}
__device__ inline void wait_async0() {
  asm volatile("s_wait_asynccnt 0" ::: "memory");
}

// ============================ Pass 1a: RoPE Q & K -> f16 ============================
__global__ __launch_bounds__(128)
void rope_stage_kernel(const float* __restrict__ Q, const float* __restrict__ K,
                       _Float16* __restrict__ Qr, _Float16* __restrict__ Kr, int S)
{
  const int s = blockIdx.x;
  const int d = threadIdx.x;
  const int i = d & 63;
  const float inv_freq = __powf(10000.0f, -(float)i * (1.0f / 64.0f));
  float sn, cs;
  __sincosf((float)s * inv_freq, &sn, &cs);
  const float sgn = (d < 64) ? -1.0f : 1.0f;
  const int dp = d ^ 64;
  const float scale = 0.08838834764831845f;  // 1/sqrt(128), folded into Q

  const float* qrow = Q + (size_t)s * (NH * DH);
  _Float16*    qout = Qr + (size_t)s * (NH * DH);
#pragma unroll
  for (int h = 0; h < NH; ++h) {
    float x  = qrow[h * DH + d];
    float xr = sgn * qrow[h * DH + dp];
    qout[h * DH + d] = (_Float16)((x * cs + xr * sn) * scale);
  }
  const float* krow = K + (size_t)s * (NKV * DH);
#pragma unroll
  for (int hk = 0; hk < NKV; ++hk) {
    float x  = krow[hk * DH + d];
    float xr = sgn * krow[hk * DH + dp];
    Kr[((size_t)hk * S + s) * DH + d] = (_Float16)(x * cs + xr * sn);
  }
}

// ================== Pass 1b: V -> f16 transposed [hkv][d][s] ==================
__global__ __launch_bounds__(128)
void vtr_kernel(const float* __restrict__ V, _Float16* __restrict__ Vr, int S)
{
  const int d  = threadIdx.x;
  const int hk = blockIdx.y;
  const int s0 = blockIdx.x * 8;
  v8h out;
#pragma unroll
  for (int j = 0; j < 8; ++j)
    out[j] = (_Float16)V[(size_t)(s0 + j) * (NKV * DH) + hk * DH + d];
  *(v8h*)&Vr[((size_t)hk * DH + d) * S + s0] = out;
}

// ============ Pass 2: flash attention, double-buffered async staging ============
__global__ __launch_bounds__(128)
void fa_kernel(const _Float16* __restrict__ Qr, const _Float16* __restrict__ Kr,
               const _Float16* __restrict__ Vr, const float* __restrict__ SNK,
               const int* __restrict__ WSZ, float* __restrict__ OUT, int S)
{
  __shared__ _Float16 Ks[2][32][KP];    // roped K chunk, [key][d]
  __shared__ _Float16 Vt[2][DH][VP];    // V chunk transposed [d][key]
  __shared__ _Float16 Ps[GQ][QT][VP];   // probs staging per wave

  const int tid  = threadIdx.x;
  const int lane = tid & 31;
  const int wv   = tid >> 5;
  const int qb   = blockIdx.x * QT;
  const int hkv  = blockIdx.y;
  const int h    = hkv * GQ + wv;
  const int win  = WSZ[0];

  const _Float16* kbase = Kr + (size_t)hkv * S * DH;
  const _Float16* vbase = Vr + (size_t)hkv * DH * S;

  // ---- Q A-tiles straight from global (roped, pre-scaled f16) ----
  AV qa[4];
  {
    const int m  = lane & 15;
    const int ko = (lane < 16) ? 0 : 8;
    const _Float16* qrow = Qr + ((size_t)(qb + m) * NH + h) * DH;
#pragma unroll
    for (int c = 0; c < 4; ++c) {
      qa[c].h[0] = *(const v8h*)&qrow[32 * c + ko];
      qa[c].h[1] = *(const v8h*)&qrow[32 * c + 16 + ko];
    }
  }

  float mrow[8], lrow[8], alpha[8];
  v8f o[8];
  const float sink = SNK[h];
#pragma unroll
  for (int r = 0; r < 8; ++r) { mrow[r] = sink; lrow[r] = 1.0f; }
#pragma unroll
  for (int c = 0; c < 8; ++c) o[c] = (v8f){};

  const int t_end    = qb + QT - 1;
  const int t_begin  = max(0, qb - (win - 1));
  const int cb0      = (t_begin >> 5) << 5;
  const int mrow_off = (lane < 16) ? 0 : 8;

  // per-thread staging geometry (uniform shape every chunk)
  const int kr = tid >> 2;            // K row 0..31
  const int kc = (tid & 3) * 32;      // 32-half column segment

  auto stage = [&](int buf, int cb) {
    async_cp4((unsigned)(uintptr_t)&Ks[buf][kr][kc], kbase,
              ((cb + kr) * DH + kc) * 2);
    async_cp4((unsigned)(uintptr_t)&Vt[buf][tid][0], vbase,
              (tid * S + cb) * 2);
  };

  int buf = 0;
  stage(0, cb0);  // prologue

  for (int cb = cb0; cb <= t_end; cb += 32) {
    wait_async0();
    __syncthreads();                       // all waves' chunk data in LDS; prev compute done
    if (cb + 32 <= t_end) stage(buf ^ 1, cb + 32);   // overlap next copy with compute

    // ---- scores: two 16x16 subtiles, K-dim = 128 via 4 WMMAs each ----
    v8f sc[2];
#pragma unroll
    for (int sub = 0; sub < 2; ++sub) {
      v8f a = (v8f){};
      const int key = sub * 16 + (lane & 15);
      const int dko = (lane < 16) ? 0 : 8;
#pragma unroll
      for (int c = 0; c < 4; ++c) {
        AV b;
        b.h[0] = *(const v8h*)&Ks[buf][key][32 * c + dko];
        b.h[1] = *(const v8h*)&Ks[buf][key][32 * c + dko + 16];
        a = __builtin_amdgcn_wmma_f32_16x16x32_f16(false, qa[c].v, false, b.v,
                                                   (short)0, a, false, false);
      }
      sc[sub] = a;
    }

    // ---- mask only on boundary chunks (uniform branch) ----
    const bool interior = (cb + 31 <= qb) && (cb >= qb + QT - win);
    if (!interior) {
#pragma unroll
      for (int sub = 0; sub < 2; ++sub) {
        const int t = cb + sub * 16 + (lane & 15);
#pragma unroll
        for (int r = 0; r < 8; ++r) {
          const int s = qb + r + mrow_off;
          const bool ok = (t <= s) && (s - t < win);
          sc[sub][r] = ok ? sc[sub][r] : -1e30f;
        }
      }
    }

    // ---- online softmax update (DPP16 reductions) ----
#pragma unroll
    for (int r = 0; r < 8; ++r) {
      float mx = redmax16(fmaxf(sc[0][r], sc[1][r]));
      const float mnew = fmaxf(mrow[r], mx);
      alpha[r] = __expf(mrow[r] - mnew);
      mrow[r] = mnew;
      const float p0 = __expf(sc[0][r] - mnew);
      const float p1 = __expf(sc[1][r] - mnew);
      sc[0][r] = p0; sc[1][r] = p1;
      lrow[r] = lrow[r] * alpha[r] + redsum16(p0 + p1);
    }

    // ---- P: C-layout f32 -> row-major f16 in LDS (wave-private buffer) ----
#pragma unroll
    for (int sub = 0; sub < 2; ++sub)
#pragma unroll
      for (int r = 0; r < 8; ++r)
        Ps[wv][r + mrow_off][sub * 16 + (lane & 15)] = (_Float16)sc[sub][r];
    asm volatile("s_wait_dscnt 0" ::: "memory");

    AV pa;
    {
      const int m  = lane & 15;
      const int ko = (lane < 16) ? 0 : 8;
      pa.h[0] = *(const v8h*)&Ps[wv][m][ko];
      pa.h[1] = *(const v8h*)&Ps[wv][m][16 + ko];
    }

    // ---- rescale accumulators, then PV ----
#pragma unroll
    for (int c = 0; c < 8; ++c)
#pragma unroll
      for (int r = 0; r < 8; ++r)
        o[c][r] *= alpha[r];
#pragma unroll
    for (int c = 0; c < 8; ++c) {
      AV b;
      const int dcol = 16 * c + (lane & 15);
      const int kko  = (lane < 16) ? 0 : 8;
      b.h[0] = *(const v8h*)&Vt[buf][dcol][kko];
      b.h[1] = *(const v8h*)&Vt[buf][dcol][kko + 16];
      o[c] = __builtin_amdgcn_wmma_f32_16x16x32_f16(false, pa.v, false, b.v,
                                                    (short)0, o[c], false, false);
    }
    buf ^= 1;
  }

  // ---- finalize ----
  {
    const int n = lane & 15;
#pragma unroll
    for (int r = 0; r < 8; ++r) {
      const float inv = 1.0f / lrow[r];
      const int s = qb + r + mrow_off;
      float* orow = OUT + (size_t)s * (NH * DH) + (size_t)h * DH;
#pragma unroll
      for (int c = 0; c < 8; ++c)
        orow[16 * c + n] = o[c][r] * inv;
    }
  }
}

// ===================== Fallback: single-pass kernel (ws too small) =====================
__device__ inline float rope_val(const float* __restrict__ row, int d, float pos) {
  int i = d & 63;
  float inv_freq = __powf(10000.0f, -(float)i * (1.0f / 64.0f));
  float ang = pos * inv_freq;
  float s, c;
  __sincosf(ang, &s, &c);
  float x  = row[d];
  float xr = (d < 64) ? -row[d + 64] : row[d - 64];
  return x * c + xr * s;
}

__global__ __launch_bounds__(128)
void fa_onepass_kernel(const float* __restrict__ Q, const float* __restrict__ K,
                       const float* __restrict__ V, const float* __restrict__ SNK,
                       const int* __restrict__ WSZ, float* __restrict__ OUT, int S)
{
  __shared__ _Float16 Qs[GQ][QT][KP];
  __shared__ _Float16 Ks[32][KP];
  __shared__ _Float16 Vt[DH][VP];
  __shared__ _Float16 Ps[GQ][QT][VP];

  const int tid  = threadIdx.x;
  const int lane = tid & 31;
  const int wv   = tid >> 5;
  const int qb   = blockIdx.x * QT;
  const int hkv  = blockIdx.y;
  const int h    = hkv * GQ + wv;
  const int win  = WSZ[0];

  {
    const int r  = tid >> 3;
    const int d0 = (tid & 7) * 16;
    const float pos = (float)(qb + r);
#pragma unroll
    for (int g = 0; g < GQ; ++g) {
      const float* qrow = Q + (size_t)(qb + r) * (NH * DH) + (size_t)(hkv * GQ + g) * DH;
      for (int d = d0; d < d0 + 16; ++d)
        Qs[g][r][d] = (_Float16)(rope_val(qrow, d, pos) * 0.08838834764831845f);
    }
  }
  __syncthreads();

  AV qa[4];
  {
    const int m  = lane & 15;
    const int ko = (lane < 16) ? 0 : 8;
#pragma unroll
    for (int c = 0; c < 4; ++c) {
      qa[c].h[0] = *(const v8h*)&Qs[wv][m][32 * c + ko];
      qa[c].h[1] = *(const v8h*)&Qs[wv][m][32 * c + 16 + ko];
    }
  }

  float mrow[8], lrow[8], alpha[8];
  v8f o[8];
  const float sink = SNK[h];
#pragma unroll
  for (int r = 0; r < 8; ++r) { mrow[r] = sink; lrow[r] = 1.0f; }
#pragma unroll
  for (int c = 0; c < 8; ++c) o[c] = (v8f){};

  const int t_end    = qb + QT - 1;
  const int t_begin  = max(0, qb - (win - 1));
  const int cb0      = (t_begin >> 5) << 5;
  const int mrow_off = (lane < 16) ? 0 : 8;

  for (int cb = cb0; cb <= t_end; cb += 32) {
    __syncthreads();
    {
      const int r  = tid >> 2;
      const int d0 = (tid & 3) * 32;
      const int t  = cb + r;
      const float* krow = K + (size_t)t * (NKV * DH) + (size_t)hkv * DH;
      const float* vrow = V + (size_t)t * (NKV * DH) + (size_t)hkv * DH;
      const float pos = (float)t;
      for (int d = d0; d < d0 + 32; ++d) {
        Ks[r][d] = (_Float16)rope_val(krow, d, pos);
        Vt[d][r] = (_Float16)vrow[d];
      }
    }
    __syncthreads();

    v8f sc[2];
#pragma unroll
    for (int sub = 0; sub < 2; ++sub) {
      v8f a = (v8f){};
      const int key = sub * 16 + (lane & 15);
      const int dko = (lane < 16) ? 0 : 8;
#pragma unroll
      for (int c = 0; c < 4; ++c) {
        AV b;
        b.h[0] = *(const v8h*)&Ks[key][32 * c + dko];
        b.h[1] = *(const v8h*)&Ks[key][32 * c + dko + 16];
        a = __builtin_amdgcn_wmma_f32_16x16x32_f16(false, qa[c].v, false, b.v,
                                                   (short)0, a, false, false);
      }
      sc[sub] = a;
    }

    const bool interior = (cb + 31 <= qb) && (cb >= qb + QT - win);
    if (!interior) {
#pragma unroll
      for (int sub = 0; sub < 2; ++sub) {
        const int t = cb + sub * 16 + (lane & 15);
#pragma unroll
        for (int r = 0; r < 8; ++r) {
          const int s = qb + r + mrow_off;
          const bool ok = (t <= s) && (s - t < win);
          sc[sub][r] = ok ? sc[sub][r] : -1e30f;
        }
      }
    }

#pragma unroll
    for (int r = 0; r < 8; ++r) {
      float mx = redmax16(fmaxf(sc[0][r], sc[1][r]));
      const float mnew = fmaxf(mrow[r], mx);
      alpha[r] = __expf(mrow[r] - mnew);
      mrow[r] = mnew;
      const float p0 = __expf(sc[0][r] - mnew);
      const float p1 = __expf(sc[1][r] - mnew);
      sc[0][r] = p0; sc[1][r] = p1;
      lrow[r] = lrow[r] * alpha[r] + redsum16(p0 + p1);
    }

#pragma unroll
    for (int sub = 0; sub < 2; ++sub)
#pragma unroll
      for (int r = 0; r < 8; ++r)
        Ps[wv][r + mrow_off][sub * 16 + (lane & 15)] = (_Float16)sc[sub][r];
    asm volatile("s_wait_dscnt 0" ::: "memory");

    AV pa;
    {
      const int m  = lane & 15;
      const int ko = (lane < 16) ? 0 : 8;
      pa.h[0] = *(const v8h*)&Ps[wv][m][ko];
      pa.h[1] = *(const v8h*)&Ps[wv][m][16 + ko];
    }

#pragma unroll
    for (int c = 0; c < 8; ++c)
#pragma unroll
      for (int r = 0; r < 8; ++r)
        o[c][r] *= alpha[r];
#pragma unroll
    for (int c = 0; c < 8; ++c) {
      AV b;
      const int dcol = 16 * c + (lane & 15);
      const int kko  = (lane < 16) ? 0 : 8;
      b.h[0] = *(const v8h*)&Vt[dcol][kko];
      b.h[1] = *(const v8h*)&Vt[dcol][kko + 16];
      o[c] = __builtin_amdgcn_wmma_f32_16x16x32_f16(false, pa.v, false, b.v,
                                                    (short)0, o[c], false, false);
    }
  }

  {
    const int n = lane & 15;
#pragma unroll
    for (int r = 0; r < 8; ++r) {
      const float inv = 1.0f / lrow[r];
      const int s = qb + r + mrow_off;
      float* orow = OUT + (size_t)s * (NH * DH) + (size_t)h * DH;
#pragma unroll
      for (int c = 0; c < 8; ++c)
        orow[16 * c + n] = o[c][r] * inv;
    }
  }
}

// ============================ Host launcher ============================
extern "C" void kernel_launch(void* const* d_in, const int* in_sizes, int n_in,
                              void* d_out, int out_size, void* d_ws, size_t ws_size,
                              hipStream_t stream) {
  const float* q   = (const float*)d_in[0];
  const float* k   = (const float*)d_in[1];
  const float* v   = (const float*)d_in[2];
  const float* snk = (const float*)d_in[3];
  const int*   wsz = (const int*)d_in[4];
  float* out = (float*)d_out;

  const int S = in_sizes[0] / (NH * DH);   // 2048

  const size_t qbytes = (size_t)S * NH * DH * sizeof(_Float16);
  const size_t kbytes = (size_t)NKV * S * DH * sizeof(_Float16);
  const size_t need   = qbytes + 2 * kbytes;

  if (ws_size >= need) {
    _Float16* Qr = (_Float16*)d_ws;
    _Float16* Kr = (_Float16*)((char*)d_ws + qbytes);
    _Float16* Vr = (_Float16*)((char*)d_ws + qbytes + kbytes);

    rope_stage_kernel<<<dim3(S), 128, 0, stream>>>(q, k, Qr, Kr, S);
    vtr_kernel<<<dim3(S / 8, NKV), 128, 0, stream>>>(v, Vr, S);
    fa_kernel<<<dim3(S / QT, NKV), 128, 0, stream>>>(Qr, Kr, Vr, snk, wsz, out, S);
  } else {
    fa_onepass_kernel<<<dim3(S / QT, NKV), 128, 0, stream>>>(q, k, v, snk, wsz, out, S);
  }
}
